// HANNA_Ensemble_Multicomponent_42099269435975
// MI455X (gfx1250) — matrix operand order repaired and checked by reading, not compiled, not run on hardware
//
#include <hip/hip_runtime.h>
#include <hip/hip_bf16.h>

// ---------------- types ----------------
typedef __attribute__((ext_vector_type(16))) __bf16 bf16x16;
typedef __attribute__((ext_vector_type(8)))  __bf16 bf16x8;
typedef __attribute__((ext_vector_type(8)))  float  floatx8;

// ---------------- problem constants ----------------
constexpr int Bsz   = 16384;
constexpr int Ncomp = 5;
constexpr int Ddim  = 384;
constexpr int NODES = 96;
constexpr int MMOD  = 3;
constexpr int NPAIR = 10;

// packed-weight fragment block: (kt,cb) -> 32 lanes x 16 elems
constexpr int PK0 = 12 * 6 * 512;  // layer0: K=384 -> 12 kt blocks
constexpr int PK1 = 4  * 6 * 512;  // layer1: K=98 pad 128 -> 4 kt
constexpr int PK2 = 3  * 6 * 512;  // 96x96 layers

__constant__ int PI_c[10] = {0,0,0,0,1,1,1,2,2,3};
__constant__ int PJ_c[10] = {1,2,3,4,2,3,4,3,4,4};
// index of unordered pair (i,j) into the 15-dot list
__constant__ int DOTIX_c[25] = {
  0, 1, 2, 3, 4,
  1, 5, 6, 7, 8,
  2, 6, 9,10,11,
  3, 7,10,12,13,
  4, 8,11,13,14 };
__constant__ int DI_c[15] = {0,0,0,0,0,1,1,1,1,2,2,2,3,3,4};
__constant__ int DJ_c[15] = {0,1,2,3,4,1,2,3,4,2,3,4,3,4,4};

// ---------------- workspace layout ----------------
struct WSL {
  __bf16 *pk0, *pk1, *pk2, *pk3, *pk2b, *pk3b;
  float  *bias;   // [M][5][96]
  float  *w4;     // [M][96]
  float  *w1c;    // [M][96]  (column 96 of Ws1)
  __bf16 *Ebf;    // [B*5][384]  bf16 copy of E (converted once)
  float  *th;     // [B*5][96]  f32 (for pairwise/rbf precision)
  __bf16 *thb;    // [B*5][96]  bf16 (for c-tile builds)
  float  *x;      // [B][5]
  float  *rbf;    // [B][25]
  float  *Xi, *Xj, *corr;  // [B][10]
  float  *dx;     // [M][B][5] grad accumulators
  float  *ge;     // [M][B]
};

__host__ __device__ inline WSL make_ws(char* base) {
  size_t off = 0;
  auto take = [&](size_t bytes) -> char* {
    char* p = base + off;
    off = (off + bytes + 255) & ~(size_t)255;
    return p;
  };
  WSL s;
  s.pk0  = (__bf16*)take((size_t)MMOD * PK0 * 2);
  s.pk1  = (__bf16*)take((size_t)MMOD * PK1 * 2);
  s.pk2  = (__bf16*)take((size_t)MMOD * PK2 * 2);
  s.pk3  = (__bf16*)take((size_t)MMOD * PK2 * 2);
  s.pk2b = (__bf16*)take((size_t)MMOD * PK2 * 2);
  s.pk3b = (__bf16*)take((size_t)MMOD * PK2 * 2);
  s.bias = (float*)take((size_t)MMOD * 5 * 96 * 4);
  s.w4   = (float*)take((size_t)MMOD * 96 * 4);
  s.w1c  = (float*)take((size_t)MMOD * 96 * 4);
  s.Ebf  = (__bf16*)take((size_t)Bsz * Ncomp * Ddim * 2);
  s.th   = (float*)take((size_t)Bsz * Ncomp * NODES * 4);
  s.thb  = (__bf16*)take((size_t)Bsz * Ncomp * NODES * 2);
  s.x    = (float*)take((size_t)Bsz * Ncomp * 4);
  s.rbf  = (float*)take((size_t)Bsz * 25 * 4);
  s.Xi   = (float*)take((size_t)Bsz * NPAIR * 4);
  s.Xj   = (float*)take((size_t)Bsz * NPAIR * 4);
  s.corr = (float*)take((size_t)Bsz * NPAIR * 4);
  s.dx   = (float*)take((size_t)MMOD * Bsz * Ncomp * 4);
  s.ge   = (float*)take((size_t)MMOD * Bsz * 4);
  return s;
}

// ---------------- WMMA helpers ----------------
// A-frag (16x32 bf16): lane L (g=L>>4, m=L&15); e<8 -> K=g*8+e ; e>=8 -> K=16+g*8+(e-8)
__device__ inline bf16x16 afrag_from_bf16(const __bf16* p) {
  bf16x8 lo = *(const bf16x8*)p;
  bf16x8 hi = *(const bf16x8*)(p + 16);
  bf16x16 A;
#pragma unroll
  for (int e = 0; e < 8; ++e) { A[e] = lo[e]; A[8 + e] = hi[e]; }
  return A;
}

// 16 x (KT*32) LDS tile (ldk=128) times packed weights -> 16x96 accumulators
__device__ inline void gemm_tile(const __bf16* tile, const __bf16* pk, int KT,
                                 floatx8 acc[6], int lane) {
  int g = lane >> 4, mm = lane & 15;
#pragma unroll
  for (int i = 0; i < 6; ++i)
#pragma unroll
    for (int e = 0; e < 8; ++e) acc[i][e] = 0.f;
  for (int kt = 0; kt < KT; ++kt) {
    bf16x16 A = afrag_from_bf16(tile + mm * 128 + kt * 32 + g * 8);
#pragma unroll
    for (int cb = 0; cb < 6; ++cb) {
      bf16x16 Bf = *(const bf16x16*)(pk + ((size_t)(kt * 6 + cb) * 512 + lane * 16));
      acc[cb] = __builtin_amdgcn_wmma_f32_16x16x32_bf16(
          false, A, false, Bf, (short)0, acc[cb], false, false);
    }
  }
}

// D-layout: lane (g=L>>4, n=L&15), vgpr r -> element (M=g*8+r, N=cb*16+n)
__device__ inline void epi_silu_stage(const floatx8 acc[6], const float* bias,
                                      __bf16* stageOut, __bf16* sp, int lane) {
  int g = lane >> 4, n = lane & 15;
#pragma unroll
  for (int cb = 0; cb < 6; ++cb) {
    int col = cb * 16 + n;
    float bv = bias[col];
#pragma unroll
    for (int r = 0; r < 8; ++r) {
      int m = g * 8 + r;
      float z = acc[cb][r] + bv;
      float s = 1.f / (1.f + __expf(-z));
      stageOut[m * 128 + col] = (__bf16)(z * s);
      sp[m * 96 + col] = (__bf16)(s * (1.f + z * (1.f - s)));
    }
  }
}

__device__ inline void epi_silu_acc_a(const floatx8 acc[6], const float* bias,
                                      __bf16* sp, float* a_t, bool add, int lane) {
  int g = lane >> 4, n = lane & 15;
#pragma unroll
  for (int cb = 0; cb < 6; ++cb) {
    int col = cb * 16 + n;
    float bv = bias[col];
#pragma unroll
    for (int r = 0; r < 8; ++r) {
      int m = g * 8 + r;
      float z = acc[cb][r] + bv;
      float s = 1.f / (1.f + __expf(-z));
      float h = z * s;
      sp[m * 96 + col] = (__bf16)(s * (1.f + z * (1.f - s)));
      if (add) a_t[m * 96 + col] += h; else a_t[m * 96 + col] = h;
    }
  }
}

__device__ inline void epi_store_a(const floatx8 acc[6], float* a_t, int lane) {
  int g = lane >> 4, n = lane & 15;
#pragma unroll
  for (int cb = 0; cb < 6; ++cb) {
    int col = cb * 16 + n;
#pragma unroll
    for (int r = 0; r < 8; ++r) a_t[(g * 8 + r) * 96 + col] = acc[cb][r];
  }
}

// gx[m] = sum_col (acc(dh1)[m,col] * sp1[m,col]) * w1c[col]   (shfl-xor over the 16 N-lanes)
__device__ inline void reduce_gx(const floatx8 acc[6], const __bf16* sp1,
                                 const float* w1c, float* gx_out, int lane) {
  int g = lane >> 4, n = lane & 15;
  float sums[8];
#pragma unroll
  for (int r = 0; r < 8; ++r) sums[r] = 0.f;
#pragma unroll
  for (int cb = 0; cb < 6; ++cb) {
    int col = cb * 16 + n;
    float wc = w1c[col];
#pragma unroll
    for (int r = 0; r < 8; ++r) {
      int m = g * 8 + r;
      sums[r] += acc[cb][r] * (float)sp1[m * 96 + col] * wc;
    }
  }
#pragma unroll
  for (int mask = 1; mask < 16; mask <<= 1)
#pragma unroll
    for (int r = 0; r < 8; ++r) sums[r] += __shfl_xor(sums[r], mask, 32);
#pragma unroll
  for (int r = 0; r < 8; ++r)
    if ((lane & 15) == r) gx_out[g * 8 + r] = sums[r];
}

// ---------------- E -> bf16 (once per launch) ----------------
__global__ __launch_bounds__(256) void convert_E_kernel(const float* E, char* ws) {
  WSL S = make_ws(ws);
  size_t t = ((size_t)blockIdx.x * 256 + threadIdx.x) * 8;  // 8 floats per thread
  float4 a0 = *(const float4*)(E + t);
  float4 a1 = *(const float4*)(E + t + 4);
  bf16x8 o;
  o[0]=(__bf16)a0.x; o[1]=(__bf16)a0.y; o[2]=(__bf16)a0.z; o[3]=(__bf16)a0.w;
  o[4]=(__bf16)a1.x; o[5]=(__bf16)a1.y; o[6]=(__bf16)a1.z; o[7]=(__bf16)a1.w;
  *(bf16x8*)(S.Ebf + t) = o;
}

// ---------------- prep: spectral-normalize + pack weights ----------------
struct PrepArgs {
  const float* W[5]; const float* b[5]; const float* ci[5];
  const float* u[5]; const float* v[5];
};

__global__ __launch_bounds__(256) void prep_kernel(PrepArgs pa, char* ws) {
  WSL S = make_ws(ws);
  int blk = blockIdx.x, m = blk / 5, l = blk % 5, tid = threadIdx.x;
  const int FIN[5]  = {384, 98, 96, 96, 96};
  const int FOUT[5] = {96, 96, 96, 96, 1};
  int fin = FIN[l], fout = FOUT[l];
  const float* Wm = pa.W[l] + (size_t)m * fout * fin;
  const float* bm = pa.b[l] + m * fout;
  const float* um = pa.u[l] + m * fout;
  const float* vm = pa.v[l] + m * fin;
  float civ = pa.ci[l][m];

  __shared__ float red[256];
  float part = 0.f;
  for (int o = tid; o < fout; o += 256) {
    float wv = 0.f;
    for (int i2 = 0; i2 < fin; ++i2) wv += Wm[(size_t)o * fin + i2] * vm[i2];
    part += um[o] * wv;
  }
  red[tid] = part; __syncthreads();
  for (int s2 = 128; s2 > 0; s2 >>= 1) {
    if (tid < s2) red[tid] += red[tid + s2];
    __syncthreads();
  }
  float sv = red[0];
  float scale = logf(1.f + expf(civ)) / (sv + 1e-12f);

  for (int o = tid; o < fout; o += 256) S.bias[(m * 5 + l) * 96 + o] = bm[o];
  if (l == 4) for (int c = tid; c < 96; c += 256) S.w4[m * 96 + c] = Wm[c] * scale;
  if (l == 1) for (int o = tid; o < 96; o += 256) S.w1c[m * 96 + o] = Wm[(size_t)o * 98 + 96] * scale;

  if (l < 4) {  // forward pack: B[k][n] = Ws[n][k]
    const int KT[4] = {12, 4, 3, 3};
    __bf16* dst = (l == 0) ? S.pk0 + (size_t)m * PK0
                : (l == 1) ? S.pk1 + (size_t)m * PK1
                : (l == 2) ? S.pk2 + (size_t)m * PK2
                           : S.pk3 + (size_t)m * PK2;
    int elems = KT[l] * 6 * 512;
    for (int t = tid; t < elems; t += 256) {
      int e = t & 15, L = (t >> 4) & 31, cbkt = t >> 9, cb = cbkt % 6, kt = cbkt / 6;
      int n = cb * 16 + (L & 15);
      int k = kt * 32 + ((e < 8) ? ((L >> 4) * 8 + e) : (16 + (L >> 4) * 8 + (e - 8)));
      float val = (k < fin) ? Wm[(size_t)n * fin + k] * scale : 0.f;
      dst[t] = (__bf16)val;
    }
  }
  if (l == 2 || l == 3) {  // backward pack: B[k][n] = Ws[k][n]
    __bf16* dst = ((l == 2) ? S.pk2b : S.pk3b) + (size_t)m * PK2;
    for (int t = tid; t < PK2; t += 256) {
      int e = t & 15, L = (t >> 4) & 31, cbkt = t >> 9, cb = cbkt % 6, kt = cbkt / 6;
      int n = cb * 16 + (L & 15);
      int k = kt * 32 + ((e < 8) ? ((L >> 4) * 8 + e) : (16 + (L >> 4) * 8 + (e - 8)));
      dst[t] = (__bf16)(Wm[(size_t)k * 96 + n] * scale);
    }
  }
}

// ---------------- layer 0: th = silu(E @ Ws0^T + b0), 32-row tiles ----------------
__global__ __launch_bounds__(256) void layer0_kernel(int model, char* ws) {
  WSL S = make_ws(ws);
  int lane = threadIdx.x & 31, w = threadIdx.x >> 5;
  long tile = blockIdx.x * 8 + w;          // 32 rows per wave
  long r0 = tile * 32;
  const __bf16* pk = S.pk0 + (size_t)model * PK0;
  const float* b0 = S.bias + (model * 5 + 0) * 96;
  int g = lane >> 4, mrow = lane & 15;
  const __bf16* arow0 = S.Ebf + (size_t)(r0 + mrow) * Ddim;
  const __bf16* arow1 = S.Ebf + (size_t)(r0 + 16 + mrow) * Ddim;
  floatx8 acc0[6], acc1[6];
#pragma unroll
  for (int i = 0; i < 6; ++i)
#pragma unroll
    for (int e = 0; e < 8; ++e) { acc0[i][e] = 0.f; acc1[i][e] = 0.f; }
#pragma unroll 1
  for (int kt = 0; kt < 12; ++kt) {
    bf16x16 A0 = afrag_from_bf16(arow0 + kt * 32 + g * 8);
    bf16x16 A1 = afrag_from_bf16(arow1 + kt * 32 + g * 8);
#pragma unroll
    for (int cb = 0; cb < 6; ++cb) {
      bf16x16 Bf = *(const bf16x16*)(pk + ((size_t)(kt * 6 + cb) * 512 + lane * 16));
      acc0[cb] = __builtin_amdgcn_wmma_f32_16x16x32_bf16(
          false, A0, false, Bf, (short)0, acc0[cb], false, false);
      acc1[cb] = __builtin_amdgcn_wmma_f32_16x16x32_bf16(
          false, A1, false, Bf, (short)0, acc1[cb], false, false);
    }
  }
#pragma unroll
  for (int half = 0; half < 2; ++half) {
    floatx8* acc = half ? acc1 : acc0;
    float*  out  = S.th  + (size_t)(r0 + half * 16) * 96;
    __bf16* outb = S.thb + (size_t)(r0 + half * 16) * 96;
#pragma unroll
    for (int cb = 0; cb < 6; ++cb) {
      int col = cb * 16 + (lane & 15);
      float bv = b0[col];
#pragma unroll
      for (int r = 0; r < 8; ++r) {
        int m = g * 8 + r;
        float z = acc[cb][r] + bv;
        float s = 1.f / (1.f + __expf(-z));
        float h = z * s;
        out[(size_t)m * 96 + col] = h;
        outb[(size_t)m * 96 + col] = (__bf16)h;
      }
    }
  }
}

// ---------------- pairwise: rbf / x_adj / Xi / Xj / corr ----------------
__global__ __launch_bounds__(256) void pairwise_kernel(const float* mf, char* ws) {
  WSL S = make_ws(ws);
  int lane = threadIdx.x & 31, w = threadIdx.x >> 5;
  long b = blockIdx.x * 8 + w;
  const float* th = S.th + (size_t)b * Ncomp * 96;
  float v[5][3];
#pragma unroll
  for (int i = 0; i < 5; ++i)
#pragma unroll
    for (int c = 0; c < 3; ++c) v[i][c] = th[i * 96 + lane + 32 * c];

  __shared__ float dots[8][15];
#pragma unroll
  for (int p = 0; p < 15; ++p) {
    int i = DI_c[p], j = DJ_c[p];
    float d = v[i][0] * v[j][0] + v[i][1] * v[j][1] + v[i][2] * v[j][2];
#pragma unroll
    for (int mask = 16; mask > 0; mask >>= 1) d += __shfl_xor(d, mask, 32);
    if (lane == 0) dots[w][p] = d;
  }
  __syncthreads();

  float xl[5]; float s4 = 0.f;
#pragma unroll
  for (int p = 0; p < 4; ++p) { xl[p] = mf[(size_t)b * 4 + p]; s4 += xl[p]; }
  xl[4] = 1.f - s4;

  auto rbfv = [&](int i, int j) {
    float d2 = dots[w][DOTIX_c[i * 5 + i]] + dots[w][DOTIX_c[j * 5 + j]]
             - 2.f * dots[w][DOTIX_c[i * 5 + j]];
    return __expf(-100.f * fmaxf(d2, 0.f));
  };

  if (lane < 25) S.rbf[(size_t)b * 25 + lane] = rbfv(lane / 5, lane % 5);
  if (lane < 5)  S.x[(size_t)b * 5 + lane] = xl[lane];
  if (lane < 10) {
    int i = PI_c[lane], j = PJ_c[lane];
    float xai = 0.f, xaj = 0.f;
#pragma unroll
    for (int t = 0; t < 5; ++t) { xai += xl[t] * rbfv(i, t); xaj += xl[t] * rbfv(j, t); }
    S.Xi[(size_t)b * 10 + lane]   = 0.5f * (1.f + xai - xaj);
    S.Xj[(size_t)b * 10 + lane]   = 0.5f * (1.f + xaj - xai);
    S.corr[(size_t)b * 10 + lane] = xl[i] * xl[j] * (1.f - rbfv(i, j));
  }
}

// ---------------- fused pair-MLP: layers 1..4 fwd + VJP ----------------
__global__ __launch_bounds__(64) void fused_mlp(int model, const float* T, char* ws) {
  WSL S = make_ws(ws);
  int lane = threadIdx.x & 31, w = threadIdx.x >> 5;
  long tileId = blockIdx.x * 2 + w;
  long r0 = tileId * 16;

  __shared__ __align__(16) __bf16 stage_s[2][16 * 128];
  __shared__ __align__(16) __bf16 sp1i_s[2][16 * 96], sp1j_s[2][16 * 96];
  __shared__ __align__(16) __bf16 sp2i_s[2][16 * 96], sp2j_s[2][16 * 96];
  __shared__ __align__(16) __bf16 sp3_s[2][16 * 96];
  __shared__ float a_s[2][16 * 96];
  __shared__ float corr_s[2][16], ge_s[2][16], gxi_s[2][16], gxj_s[2][16];
  __shared__ float w4_sh[96], w1c_sh[96];

  __bf16* stage = stage_s[w];
  __bf16 *sp1i = sp1i_s[w], *sp1j = sp1j_s[w], *sp2i = sp2i_s[w], *sp2j = sp2j_s[w], *sp3 = sp3_s[w];
  float* a_t = a_s[w];

  const __bf16* pk1  = S.pk1  + (size_t)model * PK1;
  const __bf16* pk2  = S.pk2  + (size_t)model * PK2;
  const __bf16* pk3  = S.pk3  + (size_t)model * PK2;
  const __bf16* pk2b = S.pk2b + (size_t)model * PK2;
  const __bf16* pk3b = S.pk3b + (size_t)model * PK2;
  const float* b1 = S.bias + (model * 5 + 1) * 96;
  const float* b2 = S.bias + (model * 5 + 2) * 96;
  const float* b3 = S.bias + (model * 5 + 3) * 96;
  float b4v = S.bias[(model * 5 + 4) * 96];
  float* dxm = S.dx + (size_t)model * Bsz * 5;
  float* gem = S.ge + (size_t)model * Bsz;

  for (int t = threadIdx.x; t < 96; t += 64) {
    w4_sh[t]  = S.w4[model * 96 + t];
    w1c_sh[t] = S.w1c[model * 96 + t];
  }
  if (lane < 16) {
    long row = r0 + lane; long b = row / 10; int k = (int)(row - b * 10);
    corr_s[w][lane] = S.corr[b * 10 + k];
  }
  __syncthreads();

  floatx8 acc[6];

  auto build_ctile = [&](const float* Xsel, const int* idxsel) {
    for (int t = lane; t < 16 * 128; t += 32) {
      int m = t >> 7, c = t & 127;
      long row = r0 + m; long b = row / 10; int k = (int)(row - b * 10);
      __bf16 val;
      if (c < 96)       val = S.thb[((size_t)b * 5 + idxsel[k]) * 96 + c];
      else if (c == 96) val = (__bf16)Xsel[b * 10 + k];
      else if (c == 97) val = (__bf16)T[b];
      else              val = (__bf16)0.f;
      stage[t] = val;
    }
  };

  // ---- forward, branch i ----
  build_ctile(S.Xi, PI_c);
  __syncthreads();
  gemm_tile(stage, pk1, 4, acc, lane);
  __syncthreads();
  epi_silu_stage(acc, b1, stage, sp1i, lane);
  __syncthreads();
  gemm_tile(stage, pk2, 3, acc, lane);
  __syncthreads();
  epi_silu_acc_a(acc, b2, sp2i, a_t, /*add=*/false, lane);
  __syncthreads();

  // ---- forward, branch j ----
  build_ctile(S.Xj, PJ_c);
  __syncthreads();
  gemm_tile(stage, pk1, 4, acc, lane);
  __syncthreads();
  epi_silu_stage(acc, b1, stage, sp1j, lane);
  __syncthreads();
  gemm_tile(stage, pk2, 3, acc, lane);
  __syncthreads();
  epi_silu_acc_a(acc, b2, sp2j, a_t, /*add=*/true, lane);
  __syncthreads();

  // ---- layer 3 ----
  for (int t = lane; t < 16 * 96; t += 32) {
    int m = t / 96, c = t - m * 96;
    stage[m * 128 + c] = (__bf16)a_t[t];
  }
  __syncthreads();
  gemm_tile(stage, pk3, 3, acc, lane);
  __syncthreads();
  epi_silu_stage(acc, b3, stage, sp3, lane);
  __syncthreads();

  // ---- layer 4: gE_NN = h . w4 + b4 ----
  if (lane < 16) {
    float dot = b4v;
    for (int c = 0; c < 96; ++c) dot += (float)stage[lane * 128 + c] * w4_sh[c];
    ge_s[w][lane] = dot;
  }
  __syncthreads();

  // ---- backward: dz3 = corr * w4 (x) silu'(z3) ----
  for (int t = lane; t < 16 * 96; t += 32) {
    int m = t / 96, c = t - m * 96;
    stage[m * 128 + c] = (__bf16)(corr_s[w][m] * w4_sh[c] * (float)sp3[t]);
  }
  __syncthreads();
  gemm_tile(stage, pk3b, 3, acc, lane);  // da = dz3 @ Ws3
  __syncthreads();
  epi_store_a(acc, a_t, lane);           // a_t := da
  __syncthreads();

  // branch i backward
  for (int t = lane; t < 16 * 96; t += 32) {
    int m = t / 96, c = t - m * 96;
    stage[m * 128 + c] = (__bf16)(a_t[t] * (float)sp2i[t]);
  }
  __syncthreads();
  gemm_tile(stage, pk2b, 3, acc, lane);  // dh1_i
  reduce_gx(acc, sp1i, w1c_sh, gxi_s[w], lane);
  __syncthreads();

  // branch j backward
  for (int t = lane; t < 16 * 96; t += 32) {
    int m = t / 96, c = t - m * 96;
    stage[m * 128 + c] = (__bf16)(a_t[t] * (float)sp2j[t]);
  }
  __syncthreads();
  gemm_tile(stage, pk2b, 3, acc, lane);  // dh1_j
  reduce_gx(acc, sp1j, w1c_sh, gxj_s[w], lane);
  __syncthreads();

  // ---- per-row gradient epilogue ----
  if (lane < 16) {
    long row = r0 + lane; long b = row / 10; int k = (int)(row - b * 10);
    int i = PI_c[k], j = PJ_c[k];
    float corr_v = corr_s[w][lane];
    float geNN   = ge_s[w][lane];
    float gdiff  = corr_v * (gxi_s[w][lane] - gxj_s[w][lane]);
    const float* rb = S.rbf + (size_t)b * 25;
    float rbij = rb[i * 5 + j];
    float xi_  = S.x[b * 5 + i], xj_ = S.x[b * 5 + j];
    atomicAdd(&gem[b], geNN * corr_v);
#pragma unroll
    for (int nn = 0; nn < 5; ++nn) {
      float g1 = 0.5f * (rb[i * 5 + nn] - rb[j * 5 + nn]) * gdiff;
      float g2 = geNN * (1.f - rbij) * (((nn == i) ? xj_ : 0.f) + ((nn == j) ? xi_ : 0.f));
      atomicAdd(&dxm[b * 5 + nn], g1 + g2);
    }
  }
}

// ---------------- finalize: ln_gamma / gE ensemble means ----------------
__global__ __launch_bounds__(256) void finalize_kernel(const float* mf, char* ws, float* out) {
  WSL S = make_ws(ws);
  long b = (long)blockIdx.x * 256 + threadIdx.x;
  float mfl[4];
#pragma unroll
  for (int p = 0; p < 4; ++p) mfl[p] = mf[b * 4 + p];
  float lnacc[5] = {0, 0, 0, 0, 0};
  float geacc = 0.f;
  for (int m = 0; m < MMOD; ++m) {
    const float* dxm = S.dx + (size_t)m * Bsz * 5 + (size_t)b * 5;
    float ge = S.ge[(size_t)m * Bsz + b];
    float dx4 = dxm[4];
    float s = 0.f, dg[4];
#pragma unroll
    for (int p = 0; p < 4; ++p) { dg[p] = dxm[p] - dx4; s += mfl[p] * dg[p]; }
#pragma unroll
    for (int p = 0; p < 4; ++p) lnacc[p] += ge + dg[p] - s;
    lnacc[4] += ge - s;
    geacc += ge;
  }
  const float inv = 1.f / 3.f;
#pragma unroll
  for (int n = 0; n < 5; ++n) out[b * 5 + n] = lnacc[n] * inv;
  out[(size_t)Bsz * 5 + b] = geacc * inv;
}

// ---------------- launch ----------------
extern "C" void kernel_launch(void* const* d_in, const int* in_sizes, int n_in,
                              void* d_out, int out_size, void* d_ws, size_t ws_size,
                              hipStream_t stream) {
  (void)in_sizes; (void)n_in; (void)out_size; (void)ws_size;
  const float* T  = (const float*)d_in[0];
  const float* mf = (const float*)d_in[1];
  const float* E  = (const float*)d_in[2];
  PrepArgs pa;
  for (int l = 0; l < 5; ++l) {
    pa.W[l]  = (const float*)d_in[3 + 3 * l];
    pa.b[l]  = (const float*)d_in[4 + 3 * l];
    pa.ci[l] = (const float*)d_in[5 + 3 * l];
    pa.u[l]  = (const float*)d_in[18 + 2 * l];
    pa.v[l]  = (const float*)d_in[19 + 2 * l];
  }
  char* ws = (char*)d_ws;
  WSL S = make_ws(ws);
  hipMemsetAsync(S.dx, 0, (size_t)MMOD * Bsz * 5 * sizeof(float), stream);
  hipMemsetAsync(S.ge, 0, (size_t)MMOD * Bsz * sizeof(float), stream);
  // E -> bf16 once: B*5*384 / (256 threads * 8 floats) = 15360 blocks
  convert_E_kernel<<<15360, 256, 0, stream>>>(E, ws);
  prep_kernel<<<15, 256, 0, stream>>>(pa, ws);
  for (int m = 0; m < MMOD; ++m) {
    layer0_kernel<<<320, 256, 0, stream>>>(m, ws);       // 81920 rows / (32*8)
    pairwise_kernel<<<2048, 256, 0, stream>>>(mf, ws);   // 16384 batches / 8 waves
    fused_mlp<<<5120, 64, 0, stream>>>(m, T, ws);        // 163840 rows / (16*2)
  }
  finalize_kernel<<<64, 256, 0, stream>>>(mf, ws, (float*)d_out);
}
